// TGEM_Encoder_17437567222229
// MI455X (gfx1250) — matrix operand order repaired and checked by reading, not compiled, not run on hardware
//
#include <hip/hip_runtime.h>
#include <math.h>

// Problem constants (match reference)
constexpr int Bb = 4, Nn = 2048, Hh = 5, Ll = 3;
constexpr float EPSf = 1e-6f;
#define LOG2E 1.44269504088896340736f

typedef __attribute__((ext_vector_type(16))) _Float16     v16h;
typedef __attribute__((ext_vector_type(8)))  float        v8f;
typedef __attribute__((ext_vector_type(8)))  unsigned int v8u;

// One workgroup: (b, 64-row i-tile). 128 threads = 4 wave32, each wave owns a
// 16-row WMMA tile. Loops heads h internally (no atomics). For each 32-wide j
// chunk: A[16x32] = exp2(qs_i*k_j - mm_i) in f16, B[32x16] col0=v, col1=1,
// C += A@B via v_wmma_f32_16x16x32_f16  -> C[:,0]=num, C[:,1]=den.
__global__ __launch_bounds__(128) void attn_kernel(
    const float* __restrict__ xin,   // [B,N] layer input
    const float* __restrict__ WQ,    // [H,N]
    const float* __restrict__ WK,    // [H,N]
    const float* __restrict__ WV,    // [H,N]
    const float* __restrict__ W0,    // [H]
    float* __restrict__ aout)        // [B,N] attention output (pre-LN)
{
    __shared__ float kld[Nn];                       // k = x*WK[h], f32
    __shared__ __align__(16) _Float16 vh[Nn];       // v = x*WV[h], f16
    __shared__ float qs[64];                        // q*log2e for this i-tile
    __shared__ float mm[64];                        // row max in exp2 domain
    __shared__ float red[2 * 128];                  // max/min reduction scratch
    __shared__ float ndnum[64], ndden[64];

    const int b      = blockIdx.x;
    const int itbase = blockIdx.y * 64;
    const int tid    = threadIdx.x;
    const int lane   = tid & 31;
    const int w      = tid >> 5;     // wave id 0..3
    const int m      = lane & 15;    // row-in-tile (A) / column N (B,C)
    const int kh     = lane >> 4;    // K half-select

    // B-fragment constant pack: col1 = 1.0h|1.0h, other cols (>=2) = 0
    const bool         isv   = (m == 0);
    const unsigned int cpack = (m == 1) ? 0x3C003C00u : 0u;

    const float* xrow = xin + b * Nn;
    float acc = 0.f;                 // sum_h w0[h]*out[h,i] for i = itbase+tid

    for (int h = 0; h < Hh; ++h) {
        // ---- stage k (f32) and v (f16) into LDS; partial max/min of k ----
        float pmax = -1e30f, pmin = 1e30f;
        for (int n0 = tid * 4; n0 < Nn; n0 += 128 * 4) {
            float4 xv = *(const float4*)(xrow + n0);
            float4 wk = *(const float4*)(WK + h * Nn + n0);
            float4 wv = *(const float4*)(WV + h * Nn + n0);
            float4 kv = make_float4(xv.x * wk.x, xv.y * wk.y,
                                    xv.z * wk.z, xv.w * wk.w);
            *(float4*)(kld + n0) = kv;
            vh[n0 + 0] = (_Float16)(xv.x * wv.x);
            vh[n0 + 1] = (_Float16)(xv.y * wv.y);
            vh[n0 + 2] = (_Float16)(xv.z * wv.z);
            vh[n0 + 3] = (_Float16)(xv.w * wv.w);
            pmax = fmaxf(fmaxf(fmaxf(pmax, kv.x), kv.y), fmaxf(kv.z, kv.w));
            pmin = fminf(fminf(fminf(pmin, kv.x), kv.y), fminf(kv.z, kv.w));
        }
        red[tid] = pmax; red[128 + tid] = pmin;
        __syncthreads();
        for (int s = 64; s > 0; s >>= 1) {
            if (tid < s) {
                red[tid]       = fmaxf(red[tid], red[tid + s]);
                red[128 + tid] = fminf(red[128 + tid], red[128 + tid + s]);
            }
            __syncthreads();
        }
        const float kmax = red[0], kmin = red[128];
        if (tid < 64) {
            int i = itbase + tid;
            float q = xrow[i] * WQ[h * Nn + i] * LOG2E;      // exp2-domain q
            qs[tid] = q;
            mm[tid] = (q > 0.f) ? q * kmax : q * kmin;       // max_j q*k_j
        }
        __syncthreads();

        // ---- per-wave 16-row tile, WMMA over j ----
        const int   il  = w * 16 + m;
        const float qm  = qs[il];
        const float mmr = mm[il];

        v8f c = {};
        for (int jb = 0; jb < Nn; jb += 32) {
            // A fragment: ISA 16-bit A 16x32 layout (M = lane%16)
            v16h af;
            #pragma unroll
            for (int r = 0; r < 8; ++r) {
                int K = jb + ((r >> 2) << 4) + (kh << 3) + ((r & 3) << 1);
                float e0 = __builtin_amdgcn_exp2f(qm * kld[K]     - mmr);
                float e1 = __builtin_amdgcn_exp2f(qm * kld[K + 1] - mmr);
                af[2 * r]     = (_Float16)e0;
                af[2 * r + 1] = (_Float16)e1;
            }
            // B fragment: 32x16, N = lane%16, K = (r>>2)*16 + kh*8 + 2(r&3) (+jb).
            // All 16 lanes of a half read identical LDS addresses (broadcast);
            // select between the v-pair and the constant pack -> v_cndmask.
            const int K0 = jb + (kh << 3);
            v8u bu;
            #pragma unroll
            for (int r = 0; r < 8; ++r) {
                unsigned int pv = *(const unsigned int*)&vh[K0 + ((r >> 2) << 4) + ((r & 3) << 1)];
                bu[r] = isv ? pv : cpack;
            }
            v16h bf = __builtin_bit_cast(v16h, bu);
            c = __builtin_amdgcn_wmma_f32_16x16x32_f16(
                    false, af, false, bf, (short)0, c, false, false);
        }
        // C layout: VGPR r -> row M = r + 8*kh, col N = lane%16
        if (m == 0) {
            #pragma unroll
            for (int r = 0; r < 8; ++r) ndnum[w * 16 + r + 8 * kh] = c[r];
        } else if (m == 1) {
            #pragma unroll
            for (int r = 0; r < 8; ++r) ndden[w * 16 + r + 8 * kh] = c[r];
        }
        __syncthreads();

        // ---- finalize: remove diagonal, divide, weight by w0[h] ----
        if (tid < 64) {
            int ig = itbase + tid;
            float diag = __builtin_amdgcn_exp2f(qs[tid] * kld[ig] - mm[tid])
                         * (float)vh[ig];
            float o = (ndnum[tid] - diag) / ndden[tid];
            acc += W0[h] * o;
        }
        __syncthreads();   // protect kld/vh/ndnum/ndden for next head
    }
    if (tid < 64) aout[b * Nn + itbase + tid] = acc;
}

// LayerNorm (unbiased var, /(sigma+eps)) + residual. One block per batch row.
__global__ __launch_bounds__(256) void ln_kernel(
    const float* __restrict__ prev, const float* __restrict__ a,
    const float* __restrict__ gamma, const float* __restrict__ beta,
    float* __restrict__ outp)
{
    __shared__ float s1[256], s2[256];
    const int b = blockIdx.x, tid = threadIdx.x;
    const float* ar = a + b * Nn;
    float p1 = 0.f, p2 = 0.f;
    for (int i = tid; i < Nn; i += 256) { float y = ar[i]; p1 += y; p2 += y * y; }
    s1[tid] = p1; s2[tid] = p2;
    __syncthreads();
    for (int s = 128; s > 0; s >>= 1) {
        if (tid < s) { s1[tid] += s1[tid + s]; s2[tid] += s2[tid + s]; }
        __syncthreads();
    }
    float mean = s1[0] / (float)Nn;
    float var  = (s2[0] - (float)Nn * mean * mean) / (float)(Nn - 1);
    float inv  = 1.f / (sqrtf(var) + EPSf);
    for (int i = tid; i < Nn; i += 256)
        outp[b * Nn + i] = prev[b * Nn + i] + gamma[i] * (ar[i] - mean) * inv + beta[i];
}

extern "C" void kernel_launch(void* const* d_in, const int* in_sizes, int n_in,
                              void* d_out, int out_size, void* d_ws, size_t ws_size,
                              hipStream_t stream) {
    (void)in_sizes; (void)n_in; (void)out_size; (void)ws_size;
    const float* x     = (const float*)d_in[0];   // [B,N]
    const float* WQ    = (const float*)d_in[1];   // [L,H,N]
    const float* WK    = (const float*)d_in[2];
    const float* WV    = (const float*)d_in[3];
    const float* W0    = (const float*)d_in[4];   // [L,H]
    const float* gamma = (const float*)d_in[5];   // [N]
    const float* beta  = (const float*)d_in[6];   // [N]
    float* out = (float*)d_out;

    float* abuf = (float*)d_ws;          // [B,N] attention output scratch
    float* cur  = abuf + Bb * Nn;        // [B,N] running layer output

    for (int l = 0; l < Ll; ++l) {
        const float* in = (l == 0) ? x : cur;
        float* o = (l == Ll - 1) ? out : cur;
        attn_kernel<<<dim3(Bb, Nn / 64), 128, 0, stream>>>(
            in, WQ + l * Hh * Nn, WK + l * Hh * Nn, WV + l * Hh * Nn,
            W0 + l * Hh, abuf);
        ln_kernel<<<Bb, 256, 0, stream>>>(in, abuf, gamma, beta, o);
    }
}
